// SelfAttention_83872121356936
// MI455X (gfx1250) — matrix operand order repaired and checked by reading, not compiled, not run on hardware
//
#include <hip/hip_runtime.h>
#include <hip/hip_bf16.h>

// ---------------------------------------------------------------------------
// MI455X (gfx1250) self-attention block, bf16 WMMA + async-to-LDS staging.
//   qkv = y @ W_attn + b_attn ; softmax attention ; out = O @ W_proj + b_proj
// ~137 GFLOP vs ~50 MB HBM traffic -> compute bound -> bf16 WMMA, f32 accum.
// One-time fp32->bf16 conversion (weights transposed to [N][K]) so the GEMM
// hot loops are pure {async b128 -> LDS, b32 LDS frag reads, v_wmma}.
// ---------------------------------------------------------------------------

#define Bsz 4
#define Tsz 2048
#define Csz 1024
#define Hsz 16
#define Dsz 64

typedef __attribute__((ext_vector_type(16))) __bf16 v16bf;
typedef __attribute__((ext_vector_type(8)))  float  v8f;
typedef __attribute__((ext_vector_type(4)))  int    v4i;

union Frag16 { unsigned u[8]; v16bf v; };

__device__ __forceinline__ unsigned short f32_to_bf16(float f) {
    unsigned u = __float_as_uint(f);
    u += 0x7FFFu + ((u >> 16) & 1u);   // round-to-nearest-even
    return (unsigned short)(u >> 16);
}
__device__ __forceinline__ unsigned pack2_bf16(float lo, float hi) {
    return (unsigned)f32_to_bf16(lo) | ((unsigned)f32_to_bf16(hi) << 16);
}
// A-fragment K pattern (16-bit A 16x32): VGPR v holds K = base, base+1
__device__ __forceinline__ int a_kpat(int v, int half) {
    return ((v < 4) ? 2 * v : 16 + 2 * (v - 4)) + 8 * half;
}

// --- gfx1250 async global->LDS copy (ASYNCcnt) with safe fallback ----------
#if __has_builtin(__builtin_amdgcn_global_load_async_to_lds_b128)
#define HAS_ASYNC_LDS 1
#else
#define HAS_ASYNC_LDS 0
#endif

__device__ __forceinline__ void ld_async_b128(const void* g, void* l) {
#if HAS_ASYNC_LDS
    // toolchain-probed signature: (int4* global_src, int4* lds_dst, imm, imm)
    __builtin_amdgcn_global_load_async_to_lds_b128((v4i*)g, (v4i*)l, 0, 0);
#else
    *(uint4*)l = *(const uint4*)g;
#endif
}
__device__ __forceinline__ void wait_async() {
#if HAS_ASYNC_LDS
#if __has_builtin(__builtin_amdgcn_s_wait_asynccnt)
    __builtin_amdgcn_s_wait_asynccnt(0);
#else
    asm volatile("s_wait_asynccnt 0x0" ::: "memory");
#endif
#endif
}

// ---------------------------------------------------------------------------
// One-time conversion kernels (outside the hot loops)
// ---------------------------------------------------------------------------
__global__ __launch_bounds__(256) void cvt_bf16(
    const float* __restrict__ src, unsigned short* __restrict__ dst, int npairs)
{
    const int i = blockIdx.x * 256 + threadIdx.x;
    if (i < npairs) {
        const float2 f = ((const float2*)src)[i];
        ((unsigned*)dst)[i] = pack2_bf16(f.x, f.y);
    }
}

// src[K][N] f32 -> dst[N][K] bf16 (transpose so B-fragments are contiguous)
__global__ __launch_bounds__(256) void cvt_transpose_bf16(
    const float* __restrict__ src, unsigned short* __restrict__ dst, int K, int N)
{
    const int i = blockIdx.x * 256 + threadIdx.x;
    const int kp = K >> 1;
    if (i < N * kp) {
        const int n = i / kp, k = (i % kp) * 2;
        ((unsigned*)dst)[(size_t)n * kp + (k >> 1)] =
            pack2_bf16(src[(size_t)k * N + n], src[(size_t)(k + 1) * N + n]);
    }
}

// ---------------------------------------------------------------------------
// Tiled GEMM core: 128(M)x128(N) per block, 8 waves, wave = 16 rows x 128 cols.
// Double-buffered async staging of A[128x32] and Bt[128x32] ([n][k] layout).
// ---------------------------------------------------------------------------
#define GEMM_MAIN(A_PTR, BT_PTR, KDIM)                                          \
    __shared__ unsigned short Ab[2][128 * 32];                                  \
    __shared__ unsigned short Bb[2][128 * 32];                                  \
    const int tid = threadIdx.x;                                                \
    const int lane = tid & 31, wave = tid >> 5;                                 \
    const int m0 = blockIdx.x * 128, n0 = blockIdx.y * 128;                     \
    const int half = lane >> 4, lq = lane & 15;                                 \
    v8f acc[8] = {};                                                            \
    /* stage(kk, buf): 512 b128 units each for A and Bt */                      \
    auto stage = [&](int kk, int buf) {                                         \
        _Pragma("unroll")                                                       \
        for (int u = tid; u < 512; u += 256) {                                  \
            const int r = u >> 2, c = (u & 3) * 8;                              \
            ld_async_b128(A_PTR + (size_t)(m0 + r) * KDIM + kk + c,             \
                          &Ab[buf][r * 32 + c]);                                \
            ld_async_b128(BT_PTR + (size_t)(n0 + r) * KDIM + kk + c,            \
                          &Bb[buf][r * 32 + c]);                                \
        }                                                                       \
    };                                                                          \
    stage(0, 0);                                                                \
    wait_async();                                                               \
    __syncthreads();                                                            \
    for (int kk = 0; kk < KDIM; kk += 32) {                                     \
        const int buf = (kk >> 5) & 1;                                          \
        if (kk + 32 < KDIM) stage(kk + 32, buf ^ 1);                            \
        Frag16 a;                                                               \
        _Pragma("unroll")                                                       \
        for (int v = 0; v < 8; ++v)                                             \
            a.u[v] = *(const unsigned*)&Ab[buf][(wave * 16 + lq) * 32 +         \
                                               a_kpat(v, half)];                \
        _Pragma("unroll")                                                       \
        for (int nt = 0; nt < 8; ++nt) {                                        \
            Frag16 b;                                                           \
            _Pragma("unroll")                                                   \
            for (int v = 0; v < 8; ++v)                                         \
                b.u[v] = *(const unsigned*)&Bb[buf][(nt * 16 + lq) * 32 +       \
                                                    a_kpat(v, half)];           \
            acc[nt] = __builtin_amdgcn_wmma_f32_16x16x32_bf16(                  \
                false, a.v, false, b.v, (short)0, acc[nt], false, false);       \
        }                                                                       \
        wait_async();                                                           \
        __syncthreads();                                                        \
    }

// Kernel 1: qkv = y @ W_attn + b_attn -> bf16 Q/K/V in [B,H,T,D]
__global__ __launch_bounds__(256) void qkv_gemm(
    const unsigned short* __restrict__ yb,   // [8192][1024] bf16
    const unsigned short* __restrict__ WaT,  // [3072][1024] bf16 (transposed)
    const float* __restrict__ ba,
    unsigned short* __restrict__ Qw, unsigned short* __restrict__ Kw,
    unsigned short* __restrict__ Vw)
{
    GEMM_MAIN(yb, WaT, Csz)
#pragma unroll
    for (int nt = 0; nt < 8; ++nt) {
        const int col = n0 + nt * 16 + lq;
        const float bias = ba[col];
        unsigned short* dst; int cj;
        if (col < Csz)          { dst = Qw; cj = col; }
        else if (col < 2 * Csz) { dst = Kw; cj = col - Csz; }
        else                    { dst = Vw; cj = col - 2 * Csz; }
        const int h = cj >> 6, d = cj & 63;
#pragma unroll
        for (int r = 0; r < 8; ++r) {
            const int m = m0 + wave * 16 + r + 8 * half;
            const int bb = m >> 11, t = m & (Tsz - 1);
            dst[(size_t)(((bb * Hsz + h) * Tsz) + t) * Dsz + d] =
                f32_to_bf16(acc[nt][r] + bias);
        }
    }
}

// Kernel 3: out = O(bf16) @ W_proj + b_proj (f32 output)
__global__ __launch_bounds__(256) void proj_gemm(
    const unsigned short* __restrict__ Ob,   // [8192][1024] bf16
    const unsigned short* __restrict__ WpT,  // [1024][1024] bf16 (transposed)
    const float* __restrict__ bp, float* __restrict__ out)
{
    GEMM_MAIN(Ob, WpT, Csz)
#pragma unroll
    for (int nt = 0; nt < 8; ++nt) {
        const int col = n0 + nt * 16 + lq;
        const float bias = bp[col];
#pragma unroll
        for (int r = 0; r < 8; ++r) {
            const int m = m0 + wave * 16 + r + 8 * half;
            out[(size_t)m * Csz + col] = acc[nt][r] + bias;
        }
    }
}

// ---------------------------------------------------------------------------
// Kernel 2: flash attention per (b,h). 8 waves x 16 q-rows = 128 q-rows/block.
// K tile async-staged to LDS; V cooperatively transposed; online softmax f32.
// ---------------------------------------------------------------------------
__global__ __launch_bounds__(256) void attn(
    const unsigned short* __restrict__ Qw, const unsigned short* __restrict__ Kw,
    const unsigned short* __restrict__ Vw, unsigned short* __restrict__ Ob)
{
    __shared__ unsigned short Kt[32 * 64];       // [key][d]
    __shared__ unsigned short Vt[64 * 32];       // [d][key] (transposed)
    __shared__ unsigned short Pb[8][16 * 32];    // per-wave P scratch

    const int lane = threadIdx.x & 31, wave = threadIdx.x >> 5;
    const int bh = blockIdx.x;
    const int b = bh >> 4, h = bh & 15;
    const int q0 = blockIdx.y * 128 + wave * 16;
    const int half = lane >> 4, lq = lane & 15;

    const unsigned short* Qb = Qw + (size_t)bh * Tsz * Dsz;
    const unsigned short* Kb = Kw + (size_t)bh * Tsz * Dsz;
    const unsigned short* Vb = Vw + (size_t)bh * Tsz * Dsz;

    Frag16 qa[2];
#pragma unroll
    for (int cb = 0; cb < 2; ++cb)
#pragma unroll
        for (int v = 0; v < 8; ++v) {
            const int k = cb * 32 + a_kpat(v, half);
            qa[cb].u[v] = *(const unsigned*)(Qb + (size_t)(q0 + lq) * Dsz + k);
        }

    float mrow[8], lrow[8];
#pragma unroll
    for (int r = 0; r < 8; ++r) { mrow[r] = -1e30f; lrow[r] = 0.f; }
    v8f acc[4] = {};

    for (int kt = 0; kt < Tsz; kt += 32) {
        {   // K tile: 256 async b128 units (one per thread)
            const int u = threadIdx.x;
            const int key = u >> 3, c = (u & 7) * 8;
            ld_async_b128(Kb + (size_t)(kt + key) * Dsz + c, &Kt[key * 64 + c]);
            // V tile transposed via regular loads
            const unsigned* vs = (const unsigned*)(Vb + (size_t)kt * Dsz);
            for (int i = threadIdx.x; i < 1024; i += 256) {
                const int vkey = i >> 5, dp = (i & 31) * 2;
                const unsigned pv = vs[i];
                Vt[dp * 32 + vkey]       = (unsigned short)(pv & 0xffff);
                Vt[(dp + 1) * 32 + vkey] = (unsigned short)(pv >> 16);
            }
        }
        if (kt + 32 < Tsz)
            __builtin_prefetch(Kb + (size_t)(kt + 32) * Dsz, 0, 0);
        wait_async();
        __syncthreads();

        // S = (Q K^T) * 1/sqrt(D), two 16-wide N tiles
        v8f s[2];
#pragma unroll
        for (int nt = 0; nt < 2; ++nt) {
            v8f c = {};
#pragma unroll
            for (int cb = 0; cb < 2; ++cb) {
                Frag16 bf;
#pragma unroll
                for (int v = 0; v < 8; ++v) {
                    const int d   = cb * 32 + a_kpat(v, half);
                    const int key = nt * 16 + lq;
                    bf.u[v] = *(const unsigned*)&Kt[key * 64 + d];
                }
                c = __builtin_amdgcn_wmma_f32_16x16x32_bf16(
                    false, qa[cb].v, false, bf.v, (short)0, c, false, false);
            }
            s[nt] = c * 0.125f;
        }

        // online softmax (rows striped r + 8*half across the C fragment)
#pragma unroll
        for (int r = 0; r < 8; ++r) {
            float v0 = s[0][r], v1 = s[1][r];
            float mx = fmaxf(v0, v1);
#pragma unroll
            for (int off = 8; off; off >>= 1)
                mx = fmaxf(mx, __shfl_xor(mx, off, 16));
            const float mnew  = fmaxf(mrow[r], mx);
            const float scale = __expf(mrow[r] - mnew);
            const float p0 = __expf(v0 - mnew);
            const float p1 = __expf(v1 - mnew);
            float ps = p0 + p1;
#pragma unroll
            for (int off = 8; off; off >>= 1)
                ps += __shfl_xor(ps, off, 16);
            lrow[r] = lrow[r] * scale + ps;
            mrow[r] = mnew;
#pragma unroll
            for (int nt = 0; nt < 4; ++nt) acc[nt][r] *= scale;
            const int M = r + 8 * half;
            Pb[wave][M * 32 + lq]      = f32_to_bf16(p0);
            Pb[wave][M * 32 + 16 + lq] = f32_to_bf16(p1);
        }

        // O += P @ V
        Frag16 pa;
#pragma unroll
        for (int v = 0; v < 8; ++v)
            pa.u[v] = *(const unsigned*)&Pb[wave][lq * 32 + a_kpat(v, half)];
#pragma unroll
        for (int nt = 0; nt < 4; ++nt) {
            Frag16 vb;
#pragma unroll
            for (int v = 0; v < 8; ++v) {
                const int key  = a_kpat(v, half);
                const int dcol = nt * 16 + lq;
                vb.u[v] = *(const unsigned*)&Vt[dcol * 32 + key];
            }
            acc[nt] = __builtin_amdgcn_wmma_f32_16x16x32_bf16(
                false, pa.v, false, vb.v, (short)0, acc[nt], false, false);
        }
        __syncthreads();
    }

#pragma unroll
    for (int r = 0; r < 8; ++r) {
        const float inv = 1.0f / lrow[r];
        const int t = q0 + r + 8 * half;
#pragma unroll
        for (int nt = 0; nt < 4; ++nt) {
            const int col = h * Dsz + nt * 16 + lq;
            Ob[(size_t)(b * Tsz + t) * Csz + col] = f32_to_bf16(acc[nt][r] * inv);
        }
    }
}

// ---------------------------------------------------------------------------
extern "C" void kernel_launch(void* const* d_in, const int* in_sizes, int n_in,
                              void* d_out, int out_size, void* d_ws, size_t ws_size,
                              hipStream_t stream) {
    const float* y  = (const float*)d_in[0];
    const float* Wa = (const float*)d_in[1];
    const float* ba = (const float*)d_in[2];
    const float* Wp = (const float*)d_in[3];
    const float* bp = (const float*)d_in[4];
    float* out = (float*)d_out;

    // Workspace (bf16 elements): yb 8M, WaT 3M, WpT 1M, Q/K/V 8M each, Ob 8M
    const size_t nBT = (size_t)Bsz * Tsz * Csz;          // 8M
    unsigned short* yb  = (unsigned short*)d_ws;
    unsigned short* WaT = yb  + nBT;
    unsigned short* WpT = WaT + (size_t)3 * Csz * Csz;
    unsigned short* Qw  = WpT + (size_t)Csz * Csz;
    unsigned short* Kw  = Qw + nBT;
    unsigned short* Vw  = Kw + nBT;
    unsigned short* Ob  = Vw + nBT;

    // 0) one-time conversions (y row-major; weights transposed to [N][K])
    cvt_bf16<<<(int)(nBT / 2 / 256), 256, 0, stream>>>(y, yb, (int)(nBT / 2));
    cvt_transpose_bf16<<<(3 * Csz * (Csz / 2)) / 256, 256, 0, stream>>>(
        Wa, WaT, Csz, 3 * Csz);
    cvt_transpose_bf16<<<(Csz * (Csz / 2)) / 256, 256, 0, stream>>>(
        Wp, WpT, Csz, Csz);

    // 1) QKV projection: 128x128 tiles -> grid (8192/128, 3072/128)
    qkv_gemm<<<dim3(64, 24), 256, 0, stream>>>(yb, WaT, ba, Qw, Kw, Vw);

    // 2) Attention: grid = (B*H, T/128)
    attn<<<dim3(Bsz * Hsz, Tsz / 128), 256, 0, stream>>>(Qw, Kw, Vw, Ob);

    // 3) Output projection: grid (8192/128, 1024/128)
    proj_gemm<<<dim3(64, 8), 256, 0, stream>>>(Ob, WpT, bp, out);
}
